// LearnedSpectralTokenAggregator_9904194584807
// MI455X (gfx1250) — matrix operand order repaired and checked by reading, not compiled, not run on hardware
//
#include <hip/hip_runtime.h>

typedef __attribute__((ext_vector_type(16))) __bf16 v16bf;
typedef __attribute__((ext_vector_type(8)))  float  v8f;
typedef __attribute__((ext_vector_type(2)))  float  v2f;

#define N_B   16
#define T_LEN 8192
#define D_DIM 512
#define K_Q   16
#define NEG_INF_F (-1e9f)
#define LN_EPS_F  (1e-5f)

// ---------------------------------------------------------------------------
// Phase 1: fused LayerNorm + scores = query @ x^T  (bf16 WMMA, f32 accum)
// block = 64 threads (2 waves); each wave owns 16 tokens -> one 16x16 tile
// ---------------------------------------------------------------------------
__global__ __launch_bounds__(64)
void p1_ln_scores(const float* __restrict__ tokens, const int* __restrict__ mask,
                  const float* __restrict__ query,  const float* __restrict__ lnw,
                  const float* __restrict__ lnb,    float* __restrict__ scores,
                  float* __restrict__ stats)
{
    __shared__ __bf16 xs[32][528];   // 32 tokens x 512 d (pad 16 -> 8-bank row shift)
    __shared__ __bf16 qs[16][528];   // query, bf16

    const int tid  = threadIdx.x;
    const int wave = tid >> 5, lane = tid & 31;
    const int half = lane >> 4, l16 = lane & 15;
    const int n  = blockIdx.y;
    const int t0 = blockIdx.x * 32;

    for (int i = tid; i < K_Q * D_DIM; i += 64)
        qs[i >> 9][i & 511] = (__bf16)query[i];

    float wv[16], bv[16];
    #pragma unroll
    for (int j = 0; j < 16; ++j) { wv[j] = lnw[j*32 + lane]; bv[j] = lnb[j*32 + lane]; }

    // LayerNorm: one wave handles 16 tokens, one row per iteration
    for (int i = 0; i < 16; ++i) {
        const int tt = wave*16 + i;
        const int t  = t0 + tt;
        const size_t rowbase = ((size_t)n * T_LEN + t) * D_DIM;
        float v[16]; float s = 0.f, sq = 0.f;
        #pragma unroll
        for (int j = 0; j < 16; ++j) {
            v[j] = tokens[rowbase + j*32 + lane];
            s += v[j]; sq += v[j]*v[j];
        }
        #pragma unroll
        for (int off = 16; off > 0; off >>= 1) {
            s  += __shfl_xor(s,  off, 32);
            sq += __shfl_xor(sq, off, 32);
        }
        const float mu  = s * (1.f/512.f);
        const float var = sq * (1.f/512.f) - mu*mu;
        const float rs  = rsqrtf(var + LN_EPS_F);
        #pragma unroll
        for (int j = 0; j < 16; ++j)
            xs[tt][j*32 + lane] = (__bf16)((v[j] - mu) * rs * wv[j] + bv[j]);
        if (lane == 0) {
            const size_t si = ((size_t)n*T_LEN + t) * 2;
            stats[si] = mu; stats[si+1] = rs;
        }
    }
    __syncthreads();

    // scores tile: A = query[16 x 32], B = x^T[32 x 16], K-loop over d
    v8f c = {};
    #pragma unroll
    for (int kk = 0; kk < 16; ++kk) {
        v16bf a, b;
        const int dA = kk*32 + (half ? 16 : 0);
        #pragma unroll
        for (int j = 0; j < 16; ++j) {
            a[j]   = qs[l16][dA + j];          // K = dA + j
        }
        const int dB = kk*32 + (half ? 16 : 0);
        #pragma unroll
        for (int j = 0; j < 16; ++j)
            b[j] = xs[wave*16 + l16][dB + j];  // B[K=d][N=token] = x[token][d]
        c = __builtin_amdgcn_wmma_f32_16x16x32_bf16(false, a, false, b,
                                                    (short)0, c, false, false);
    }

    const int t = t0 + wave*16 + l16;
    const int m = mask[n*T_LEN + t];
    #pragma unroll
    for (int r = 0; r < 8; ++r) {
        const int kq = half*8 + r;
        const float sc = (m == 0) ? NEG_INF_F : c[r];
        scores[((size_t)(n*K_Q + kq)) * T_LEN + t] = sc;
    }
}

// ---------------------------------------------------------------------------
// Phase 2: per-(n,k) row max and 1/sum(exp)  (deterministic tree reductions)
// ---------------------------------------------------------------------------
__global__ __launch_bounds__(256)
void p2_softmax_stats(const float* __restrict__ scores,
                      float* __restrict__ rowmax, float* __restrict__ invden)
{
    __shared__ float red[256];
    const int nk = blockIdx.x, tid = threadIdx.x;
    const float* row = scores + (size_t)nk * T_LEN;

    float m = -3.4e38f;
    for (int t = tid; t < T_LEN; t += 256) m = fmaxf(m, row[t]);
    red[tid] = m; __syncthreads();
    for (int off = 128; off > 0; off >>= 1) {
        if (tid < off) red[tid] = fmaxf(red[tid], red[tid + off]);
        __syncthreads();
    }
    m = red[0]; __syncthreads();

    float s = 0.f;
    for (int t = tid; t < T_LEN; t += 256) s += __expf(row[t] - m);
    red[tid] = s; __syncthreads();
    for (int off = 128; off > 0; off >>= 1) {
        if (tid < off) red[tid] += red[tid + off];
        __syncthreads();
    }
    if (tid == 0) { rowmax[nk] = m; invden[nk] = 1.f / red[0]; }
}

// ---------------------------------------------------------------------------
// Phase 3: attn output = exp(score - max) * invden
// ---------------------------------------------------------------------------
__global__ __launch_bounds__(256)
void p3_attn(const float* __restrict__ scores, const float* __restrict__ rowmax,
             const float* __restrict__ invden, float* __restrict__ attn)
{
    const size_t idx = (size_t)blockIdx.x * 256 + threadIdx.x;
    const int nk = (int)(idx >> 13);
    attn[idx] = __expf(scores[idx] - rowmax[nk]) * invden[nk];
}

// ---------------------------------------------------------------------------
// Phase 4: summary partials = attn @ x  (f32 WMMA 16x16x4, x recomputed from
// stored LN stats). block = 256 threads (8 waves); wave w owns d-cols [w*64, w*64+64)
// grid = (t-chunk of 512, n)
// ---------------------------------------------------------------------------
__global__ __launch_bounds__(256)
void p4_summary_partials(const float* __restrict__ tokens, const float* __restrict__ scores,
                         const float* __restrict__ stats,  const float* __restrict__ lnw,
                         const float* __restrict__ lnb,    const float* __restrict__ rowmax,
                         const float* __restrict__ invden, float* __restrict__ partials)
{
    __shared__ float xls[16][516];   // 16 tokens x 512 d, f32, padded
    __shared__ float als[16][17];    // attn tile 16(kq) x 16(t)

    const int tid  = threadIdx.x;
    const int wave = tid >> 5, lane = tid & 31;
    const int half = lane >> 4, l16 = lane & 15;
    const int n = blockIdx.y, chunk = blockIdx.x;
    const int tbase = chunk * 512;

    const int akq = tid >> 4, atc = tid & 15;
    const int ank = n*K_Q + akq;
    const float amax = rowmax[ank], ainv = invden[ank];

    const float w0 = lnw[tid], w1 = lnw[tid + 256];
    const float b0 = lnb[tid], b1 = lnb[tid + 256];

    v8f acc0 = {}, acc1 = {}, acc2 = {}, acc3 = {};

    for (int tt = 0; tt < 512; tt += 16) {
        const int t0 = tbase + tt;
        __syncthreads();   // previous iteration's WMMA reads done

        // recompute x for 16 tokens from stored LN stats (32 elems / thread)
        #pragma unroll
        for (int s = 0; s < 32; ++s) {
            const int row = s >> 1;
            const int d   = tid + (s & 1) * 256;
            const int t   = t0 + row;
            const float mu = stats[((size_t)n*T_LEN + t)*2];
            const float rs = stats[((size_t)n*T_LEN + t)*2 + 1];
            const float wv = (s & 1) ? w1 : w0;
            const float bb = (s & 1) ? b1 : b0;
            xls[row][d] = (tokens[((size_t)n*T_LEN + t)*D_DIM + d] - mu)*rs*wv + bb;
        }
        // attn tile 16x16 (one element per thread)
        als[akq][atc] = __expf(scores[(size_t)ank*T_LEN + t0 + atc] - amax) * ainv;
        __syncthreads();

        // C[16kq x 16d] += A[16kq x 4t] * B[4t x 16d], four d-tiles per wave
        #pragma unroll
        for (int kk = 0; kk < 4; ++kk) {
            v2f a;
            a.x = als[l16][kk*4 + (half ? 2 : 0)];
            a.y = als[l16][kk*4 + (half ? 3 : 1)];
            const int r0 = kk*4 + (half ? 2 : 0);
            const int r1 = kk*4 + (half ? 3 : 1);
            const int d0 = wave * 64;
            v2f b;
            b.x = xls[r0][d0 + l16];       b.y = xls[r1][d0 + l16];
            acc0 = __builtin_amdgcn_wmma_f32_16x16x4_f32(false, a, false, b, (short)0, acc0, false, false);
            b.x = xls[r0][d0 + 16 + l16];  b.y = xls[r1][d0 + 16 + l16];
            acc1 = __builtin_amdgcn_wmma_f32_16x16x4_f32(false, a, false, b, (short)0, acc1, false, false);
            b.x = xls[r0][d0 + 32 + l16];  b.y = xls[r1][d0 + 32 + l16];
            acc2 = __builtin_amdgcn_wmma_f32_16x16x4_f32(false, a, false, b, (short)0, acc2, false, false);
            b.x = xls[r0][d0 + 48 + l16];  b.y = xls[r1][d0 + 48 + l16];
            acc3 = __builtin_amdgcn_wmma_f32_16x16x4_f32(false, a, false, b, (short)0, acc3, false, false);
        }
    }

    const int d0 = wave * 64;
    #pragma unroll
    for (int r = 0; r < 8; ++r) {
        const int kq = half*8 + r;
        const size_t base = ((size_t)(n*16 + chunk)*16 + kq) * 512;
        partials[base + d0      + l16] = acc0[r];
        partials[base + d0 + 16 + l16] = acc1[r];
        partials[base + d0 + 32 + l16] = acc2[r];
        partials[base + d0 + 48 + l16] = acc3[r];
    }
}

// ---------------------------------------------------------------------------
// Phase 5: reduce 16 chunk-partials -> summary
// ---------------------------------------------------------------------------
__global__ __launch_bounds__(256)
void p5_reduce(const float* __restrict__ partials, float* __restrict__ summary)
{
    const int idx = blockIdx.x * 256 + threadIdx.x;   // 131072 total
    const int n   = idx >> 13;                        // 16*512 per n
    const int rem = idx & 8191;                       // kq*512 + d
    float s = 0.f;
    #pragma unroll
    for (int c = 0; c < 16; ++c)
        s += partials[(((size_t)(n*16 + c)) << 13) + rem];
    summary[idx] = s;
}

// ---------------------------------------------------------------------------
extern "C" void kernel_launch(void* const* d_in, const int* in_sizes, int n_in,
                              void* d_out, int out_size, void* d_ws, size_t ws_size,
                              hipStream_t stream)
{
    const float* tokens = (const float*)d_in[0];
    const int*   mask   = (const int*)  d_in[1];
    const float* query  = (const float*)d_in[2];
    const float* lnw    = (const float*)d_in[3];
    const float* lnb    = (const float*)d_in[4];

    float* out     = (float*)d_out;
    float* summary = out;                                   // [16,16,512]
    float* attn    = out + (size_t)N_B * K_Q * D_DIM;       // [16,16,8192]

    float* ws       = (float*)d_ws;
    float* scores   = ws;                                   // 16*16*8192
    float* stats    = scores + (size_t)N_B * K_Q * T_LEN;   // 16*8192*2
    float* rowmax   = stats  + (size_t)N_B * T_LEN * 2;     // 256
    float* invden   = rowmax + N_B * K_Q;                   // 256
    float* partials = invden + N_B * K_Q;                   // 16*16*16*512

    p1_ln_scores<<<dim3(T_LEN/32, N_B), 64, 0, stream>>>(tokens, mask, query, lnw, lnb,
                                                         scores, stats);
    p2_softmax_stats<<<N_B*K_Q, 256, 0, stream>>>(scores, rowmax, invden);
    p3_attn<<<(N_B*K_Q*T_LEN)/256, 256, 0, stream>>>(scores, rowmax, invden, attn);
    p4_summary_partials<<<dim3(16, N_B), 256, 0, stream>>>(tokens, scores, stats, lnw, lnb,
                                                           rowmax, invden, partials);
    p5_reduce<<<(N_B*K_Q*D_DIM)/256, 256, 0, stream>>>(partials, summary);
}
